// SoftmaxAttention_KDE_43396349559559
// MI455X (gfx1250) — matrix operand order repaired and checked by reading, not compiled, not run on hardware
//
#include <hip/hip_runtime.h>
#include <hip/hip_bf16.h>

typedef __attribute__((ext_vector_type(16))) _Float16 v16h;
typedef __attribute__((ext_vector_type(8)))  _Float16 v8h;
typedef __attribute__((ext_vector_type(2)))  _Float16 v2h;
typedef __attribute__((ext_vector_type(8)))  float    v8f;

#define BATCH 4
#define HEADS 12
#define SEQ   2048
#define DIM   64
#define BM    64          // Q rows per workgroup
#define BN    64          // K/V tokens per tile
#define LDST  72          // f16 row stride in LDS (64 + 8 pad)
#define DNORM 0.42044820762685725f            // 32^(-1/4)
#define SQRT_LOG2E 1.2011224087864498f        // sqrt(log2(e))
#define QKSCALE (DNORM * SQRT_LOG2E)          // folds log2e into Q'K'^T and diags
#define VSCALE  DNORM                         // V keeps plain normalizer
#define MBIAS2  (1.0e9f * 1.4426950408889634f) // mask bias in log2 domain

#if __has_builtin(__builtin_amdgcn_exp2f)
#define EXP2(x) __builtin_amdgcn_exp2f(x)
#else
#define EXP2(x) exp2f(x)
#endif

// ---------------------------------------------------------------------------
// WMMA fragment loads from LDS, matching CDNA5 wave32 register layouts.
// A (16x32 f16): lane L holds row M=L%16; lanes 0-15 carry K in {0..7,16..23},
// lanes 16-31 carry K in {8..15,24..31} (ISA 7.12.2). Two 16B loads + shuffle.
__device__ __forceinline__ v16h ldsA(const _Float16* buf, int row0, int k0, int lane) {
    const int m  = lane & 15;
    const int hi = (lane >> 4) & 1;
    const _Float16* p = buf + (row0 + m) * LDST + k0 + hi * 8;
    v8h lo = *(const v8h*)(p);        // K = k0 + hi*8 .. +7
    v8h up = *(const v8h*)(p + 16);   // K = k0 + 16 + hi*8 .. +7
    return __builtin_shufflevector(lo, up, 0,1,2,3,4,5,6,7,8,9,10,11,12,13,14,15);
}
// B (32x16 f16): lane L holds column N=n0+L%16; lanes 0-15 carry K in {0..7,16..23},
// lanes 16-31 carry K in {8..15,24..31} (same K order as ldsA).
__device__ __forceinline__ v16h ldsB(const _Float16* buf, int k0, int n0, int lane) {
    const int n  = lane & 15;
    const int hi = (lane >> 4) & 1;
    const _Float16* p = buf + (k0 + hi * 8) * LDST + n0 + n;
    v16h r;
    #pragma unroll
    for (int s = 0; s < 8; ++s) {
        r[s]     = p[s * LDST];
        r[s + 8] = p[(16 + s) * LDST];
    }
    return r;
}

// ---------------------------------------------------------------------------
// Pre-kernel: non_padding_num[b] = sum(mask[b, :])
__global__ void npad_kernel(const float* __restrict__ mask, float* __restrict__ out) {
    __shared__ float red[256];
    const int b = blockIdx.x;
    float s = 0.f;
    for (int i = threadIdx.x; i < SEQ; i += 256) s += mask[(size_t)b * SEQ + i];
    red[threadIdx.x] = s;
    __syncthreads();
    for (int off = 128; off > 0; off >>= 1) {
        if (threadIdx.x < off) red[threadIdx.x] += red[threadIdx.x + off];
        __syncthreads();
    }
    if (threadIdx.x == 0) out[b] = red[0];
}

// ---------------------------------------------------------------------------
// Flash-style RBF attention: one 64-row Q strip per workgroup (4 waves).
// All scores live in the log2 domain (scale folded into f16 quantization),
// so P = exp2(S) with a bare v_exp_f32 and the diag terms ride in WMMA's C.
__global__ __launch_bounds__(128) void rbf_attn_kernel(
        const float* __restrict__ Q, const float* __restrict__ K,
        const float* __restrict__ V, const float* __restrict__ mask,
        const float* __restrict__ npad, float* __restrict__ X)
{
    __shared__ __align__(32) _Float16 sQ [BM * LDST];   // A-matrix, row-major
    __shared__ __align__(32) _Float16 sKT[DIM * LDST];  // K tile transposed: sKT[k][n]
    __shared__ __align__(32) _Float16 sV [BN * LDST];   // V tile row-major: sV[k][p]
    __shared__ __align__(32) _Float16 sP [BM * LDST];   // exp2(S) strips, per wave
    __shared__ float sdQ[BM];
    __shared__ float sdK[BN];

    const int bh   = blockIdx.y;
    const int b    = bh / HEADS;
    const int q0   = blockIdx.x * BM;
    const int t    = threadIdx.x;
    const int lane = t & 31;
    const int w    = t >> 5;        // wave id 0..3 -> rows 16w..16w+15
    const int row  = t >> 1;        // 0..63: tile row this thread loads
    const int c0   = (t & 1) * 32;  // half-row columns

    const size_t base = (size_t)bh * SEQ * DIM;
    const float* Qb = Q + base;
    const float* Kb = K + base;
    const float* Vb = V + base;
    float*       Xb = X + base;
    const float* mb = mask + (size_t)b * SEQ;
    const float inv_np = 1.0f / npad[b];

    // ---- prologue: load + scale Q strip, diag_Q (log2-scaled) -----------
    {
        const int tok = q0 + row;
        const float sc = mb[tok] * QKSCALE;
        const float4* src = (const float4*)(Qb + (size_t)tok * DIM + c0);
        float ss = 0.f;
        #pragma unroll
        for (int i = 0; i < 8; ++i) {
            float4 v = src[i];
            float x0 = v.x * sc, x1 = v.y * sc, x2 = v.z * sc, x3 = v.w * sc;
            ss += x0*x0 + x1*x1 + x2*x2 + x3*x3;
            _Float16* d = &sQ[row * LDST + c0 + 4*i];
            d[0] = (_Float16)x0; d[1] = (_Float16)x1;
            d[2] = (_Float16)x2; d[3] = (_Float16)x3;
        }
        ss += __shfl_xor(ss, 1, 32);          // partner thread covers other half-row
        if ((t & 1) == 0) sdQ[row] = 0.5f * ss;
    }

    v8f acc[4] = {};                          // O strip: 16 x 64 f32 per wave

    __syncthreads();
    const v16h aq0 = ldsA(sQ, 16 * w, 0,  lane);   // Q A-fragments are loop-invariant
    const v16h aq1 = ldsA(sQ, 16 * w, 32, lane);

    // -dQ per accumulator row is invariant across the whole K loop: hoist it.
    const int hi = lane >> 4;
    float negdQ[8];
    #pragma unroll
    for (int r = 0; r < 8; ++r) negdQ[r] = -sdQ[16 * w + r + 8 * hi];

    for (int kt = 0; kt < SEQ / BN; ++kt) {
        const int tok = kt * BN + row;
        __syncthreads();                      // previous iteration done with sKT/sV/sdK

        // ---- load K tile: scale, diag_K (+mask bias), transpose to LDS --
        {
            const float mk = mb[tok];
            const float sc = mk * QKSCALE;
            const float4* src = (const float4*)(Kb + (size_t)tok * DIM + c0);
            float ss = 0.f;
            #pragma unroll
            for (int i = 0; i < 8; ++i) {
                float4 v = src[i];
                float x0 = v.x * sc, x1 = v.y * sc, x2 = v.z * sc, x3 = v.w * sc;
                ss += x0*x0 + x1*x1 + x2*x2 + x3*x3;
                const int c = c0 + 4*i;
                sKT[(c+0) * LDST + row] = (_Float16)x0;
                sKT[(c+1) * LDST + row] = (_Float16)x1;
                sKT[(c+2) * LDST + row] = (_Float16)x2;
                sKT[(c+3) * LDST + row] = (_Float16)x3;
            }
            ss += __shfl_xor(ss, 1, 32);
            if ((t & 1) == 0) sdK[row] = 0.5f * ss + MBIAS2 * (1.0f - mk);

            // ---- load V tile: scale by mask*DN/npad, row-major ----------
            const float scv = mk * VSCALE * inv_np;
            const float4* vsrc = (const float4*)(Vb + (size_t)tok * DIM + c0);
            #pragma unroll
            for (int i = 0; i < 8; ++i) {
                float4 v = vsrc[i];
                const int c = c0 + 4*i;
                v2h p0, p1;
                p0.x = (_Float16)(v.x * scv); p0.y = (_Float16)(v.y * scv);
                p1.x = (_Float16)(v.z * scv); p1.y = (_Float16)(v.w * scv);
                *(v2h*)&sV[row * LDST + c    ] = p0;
                *(v2h*)&sV[row * LDST + c + 2] = p1;
            }
        }
        if (kt + 1 < SEQ / BN) {              // global_prefetch_b8 for next tile
            const size_t ntok = (size_t)(kt + 1) * BN + row;
            __builtin_prefetch(Kb + ntok * DIM + c0, 0, 0);
            __builtin_prefetch(Vb + ntok * DIM + c0, 0, 0);
        }
        __syncthreads();

        // ---- GEMM1: S = Q'K'^T + C, C preloaded with -(dQ + dK + bias) --
        const int m0 = 16 * w;
        #pragma unroll
        for (int j = 0; j < 4; ++j) {
            const int ncol = 16*j + (lane & 15);
            const float dk = sdK[ncol];
            v8f s;
            #pragma unroll
            for (int r = 0; r < 8; ++r) s[r] = negdQ[r] - dk;
            s = __builtin_amdgcn_wmma_f32_16x16x32_f16(false, aq0, false,
                    ldsB(sKT, 0, 16*j, lane), (short)0, s, false, false);
            s = __builtin_amdgcn_wmma_f32_16x16x32_f16(false, aq1, false,
                    ldsB(sKT, 32, 16*j, lane), (short)0, s, false, false);
            #pragma unroll
            for (int r = 0; r < 8; ++r) {
                const int mloc = m0 + r + 8 * hi;
                const float pv = EXP2(s[r]);               // bare v_exp_f32
                sP[mloc * LDST + ncol] = (_Float16)pv;     // own-wave strip
            }
        }

        // ---- GEMM2: O += P Vs (same-wave LDS RAW -> no barrier needed) --
        const v16h ap0 = ldsA(sP, m0, 0,  lane);
        const v16h ap1 = ldsA(sP, m0, 32, lane);
        #pragma unroll
        for (int j = 0; j < 4; ++j) {
            acc[j] = __builtin_amdgcn_wmma_f32_16x16x32_f16(false, ap0, false,
                         ldsB(sV, 0, 16*j, lane), (short)0, acc[j], false, false);
            acc[j] = __builtin_amdgcn_wmma_f32_16x16x32_f16(false, ap1, false,
                         ldsB(sV, 32, 16*j, lane), (short)0, acc[j], false, false);
        }
    }

    // ---- epilogue: write O strip (f32), coalesced across lanes ----------
    #pragma unroll
    for (int j = 0; j < 4; ++j) {
        const int col = 16*j + (lane & 15);
        #pragma unroll
        for (int r = 0; r < 8; ++r) {
            const int orow = q0 + 16*w + r + 8 * hi;
            Xb[(size_t)orow * DIM + col] = acc[j][r];
        }
    }
}

// ---------------------------------------------------------------------------
extern "C" void kernel_launch(void* const* d_in, const int* in_sizes, int n_in,
                              void* d_out, int out_size, void* d_ws, size_t ws_size,
                              hipStream_t stream) {
    const float* Q    = (const float*)d_in[0];
    const float* K    = (const float*)d_in[1];
    const float* V    = (const float*)d_in[2];
    const float* mask = (const float*)d_in[3];
    float* npad = (float*)d_ws;              // BATCH floats of scratch
    float* X    = (float*)d_out;

    npad_kernel<<<dim3(BATCH), dim3(256), 0, stream>>>(mask, npad);

    dim3 grid(SEQ / BM, BATCH * HEADS);
    rbf_attn_kernel<<<grid, dim3(128), 0, stream>>>(Q, K, V, mask, npad, X);
}
